// SAGE_2370821947944
// MI455X (gfx1250) — matrix operand order repaired and checked by reading, not compiled, not run on hardware
//
#include <hip/hip_runtime.h>
#include <hip/hip_bf16.h>

// GraphSAGE 2-layer forward for MI455X (gfx1250, wave32, WMMA).
// Strategy: edge scatter (HBM-bound, ~1.3GB -> ~56us floor) + bf16 WMMA GEMMs
// (5 GFLOP total, trivial on WMMA units; weights staged to LDS as bf16).

typedef __bf16 bf16_t;
typedef bf16_t v16bf __attribute__((ext_vector_type(16)));
typedef float  v8f   __attribute__((ext_vector_type(8)));

#define N_NODES 50000
#define N_EDGES 640000
#define F_DIM   128   // K dim for both layers (128 input features, 128 hidden)
#define N_CLS   64

// ---------------------------------------------------------------------------
// Edge scatter: one wave per edge. Lane l handles features [4l, 4l+4).
// agg[dst] += x[src]; cnt[dst] += 1 (optional, lane 0).
// ---------------------------------------------------------------------------
__global__ void scatter_accum_kernel(const float* __restrict__ x,
                                     const int* __restrict__ src,
                                     const int* __restrict__ dst,
                                     float* __restrict__ agg,
                                     float* __restrict__ cnt,
                                     int nE) {
  int wave = blockIdx.x * (blockDim.x >> 5) + (threadIdx.x >> 5);
  int lane = threadIdx.x & 31;
  if (wave >= nE) return;
  int s = src[wave];
  int d = dst[wave];
  const float4 v = *(const float4*)(x + (size_t)s * F_DIM + lane * 4);
  float* o = agg + (size_t)d * F_DIM + lane * 4;
  atomicAdd(o + 0, v.x);
  atomicAdd(o + 1, v.y);
  atomicAdd(o + 2, v.z);
  atomicAdd(o + 3, v.w);
  if (cnt != nullptr && lane == 0) atomicAdd(cnt + d, 1.0f);
}

// inv[i] = 1 / max(cnt[i], 1)
__global__ void inv_deg_kernel(const float* __restrict__ cnt,
                               float* __restrict__ inv, int n) {
  int i = blockIdx.x * blockDim.x + threadIdx.x;
  if (i < n) inv[i] = 1.0f / fmaxf(cnt[i], 1.0f);
}

// ---------------------------------------------------------------------------
// WMMA fragment loaders (16x16x32 bf16; A and B both use contiguous k order, not the ISA table order)
// A (16x32, MxK): lane row = lane%16; g=lane/16; elem[j]=K g*16+j.
// B (32x16, KxN): lane col = lane%16; g=lane/16; elem[j]=K g*16+j.
// C/D (16x16):    lane col = lane%16; vgpr v row = v + 8*(lane/16).
// ---------------------------------------------------------------------------
__device__ __forceinline__ v16bf load_a_frag(const float* __restrict__ base,
                                             int row0, int k0, int lane,
                                             float scale) {
  int r  = row0 + (lane & 15);
  int kb = k0 + ((lane >> 4) << 4);
  const float* p = base + (size_t)r * F_DIM + kb;
  v16bf a;
#pragma unroll
  for (int e = 0; e < 8; ++e) {
    a[e]     = (bf16_t)(p[e]      * scale);
    a[e + 8] = (bf16_t)(p[e + 8] * scale);
  }
  return a;
}

__device__ __forceinline__ v16bf load_b_frag_lds(const bf16_t* __restrict__ W,
                                                 int n0, int k0, int lane) {
  // rows of W (output channels) play the role of B columns (B = W^T).
  const bf16_t* p =
      W + (size_t)(n0 + (lane & 15)) * F_DIM + k0 + ((lane >> 4) << 4);
  return *(const v16bf*)p;  // 32B contiguous, 32B-aligned -> 2x ds_load_b128
}

// ---------------------------------------------------------------------------
// Fused SAGE GEMM: out[m,:] = relu?( mean[m,:]@Wl^T + b + self[m,:]@Wr^T )
// mean[m,k] = agg[m,k] * inv[m].  NT = n_out/16 (8 for layer1, 4 for layer2).
// One wave per 16-row M-tile; A fragments reused across all NT N-tiles.
// Weights staged to LDS as bf16 once per block.
// ---------------------------------------------------------------------------
template <int NT, bool RELU>
__global__ void sage_gemm_kernel(const float* __restrict__ agg,
                                 const float* __restrict__ inv,
                                 const float* __restrict__ xself,
                                 const float* __restrict__ Wl,
                                 const float* __restrict__ Wr,
                                 const float* __restrict__ bias,
                                 float* __restrict__ out,
                                 int mtiles) {
  __shared__ __align__(16) bf16_t sWl[NT * 16 * F_DIM];
  __shared__ __align__(16) bf16_t sWr[NT * 16 * F_DIM];

  // Stage fp32 weights -> bf16 LDS (all threads, before any wave exits).
  for (int i = threadIdx.x; i < NT * 16 * F_DIM; i += blockDim.x) {
    sWl[i] = (bf16_t)Wl[i];
    sWr[i] = (bf16_t)Wr[i];
  }
  __syncthreads();

  int wave = blockIdx.x * (blockDim.x >> 5) + (threadIdx.x >> 5);
  int lane = threadIdx.x & 31;
  if (wave >= mtiles) return;
  int row0 = wave * 16;

  v8f acc[NT] = {};  // fp32 accumulators

  float sc = inv[row0 + (lane & 15)];  // per-A-row mean scale

#pragma unroll
  for (int kk = 0; kk < F_DIM; kk += 32) {
    v16bf am = load_a_frag(agg,   row0, kk, lane, sc);    // mean branch
    v16bf as = load_a_frag(xself, row0, kk, lane, 1.0f);  // self branch
#pragma unroll
    for (int t = 0; t < NT; ++t) {
      v16bf bl = load_b_frag_lds(sWl, t * 16, kk, lane);
      v16bf br = load_b_frag_lds(sWr, t * 16, kk, lane);
      acc[t] = __builtin_amdgcn_wmma_f32_16x16x32_bf16(
          false, am, false, bl, (short)0, acc[t], false, false);
      acc[t] = __builtin_amdgcn_wmma_f32_16x16x32_bf16(
          false, as, false, br, (short)0, acc[t], false, false);
    }
  }

  // Epilogue: C/D layout -> row-major store with bias (+ReLU for layer 1).
  int col16 = lane & 15;
  int moff  = (lane >> 4) * 8;
  const int ldo = NT * 16;
#pragma unroll
  for (int t = 0; t < NT; ++t) {
    int col = t * 16 + col16;
    float bv = bias[col];
#pragma unroll
    for (int v = 0; v < 8; ++v) {
      float r = acc[t][v] + bv;
      if (RELU) r = fmaxf(r, 0.0f);
      out[(size_t)(row0 + moff + v) * ldo + col] = r;
    }
  }
}

// ---------------------------------------------------------------------------
// log_softmax over 64 classes: one wave per node, 2 classes per lane.
// ---------------------------------------------------------------------------
__global__ void logsoftmax_kernel(const float* __restrict__ logits,
                                  float* __restrict__ out, int n) {
  int wave = blockIdx.x * (blockDim.x >> 5) + (threadIdx.x >> 5);
  int lane = threadIdx.x & 31;
  if (wave >= n) return;
  const float* lp = logits + (size_t)wave * N_CLS;
  float a = lp[lane];
  float b = lp[lane + 32];
  float m = fmaxf(a, b);
#pragma unroll
  for (int off = 16; off >= 1; off >>= 1) m = fmaxf(m, __shfl_xor(m, off, 32));
  float s = __expf(a - m) + __expf(b - m);
#pragma unroll
  for (int off = 16; off >= 1; off >>= 1) s += __shfl_xor(s, off, 32);
  float lse = m + __logf(s);
  float* op = out + (size_t)wave * N_CLS;
  op[lane]      = a - lse;
  op[lane + 32] = b - lse;
}

// ---------------------------------------------------------------------------
extern "C" void kernel_launch(void* const* d_in, const int* in_sizes, int n_in,
                              void* d_out, int out_size, void* d_ws,
                              size_t ws_size, hipStream_t stream) {
  const float* x   = (const float*)d_in[0];
  const int*   ei  = (const int*)d_in[1];  // [2, E] (harness passes int32)
  const float* w1l = (const float*)d_in[2];
  const float* b1l = (const float*)d_in[3];
  const float* w1r = (const float*)d_in[4];
  const float* w2l = (const float*)d_in[5];
  const float* b2l = (const float*)d_in[6];
  const float* w2r = (const float*)d_in[7];

  const int* src = ei;
  const int* dst = ei + N_EDGES;

  // Workspace carve-up (256B aligned): cnt, inv, agg1, h, agg2 (~77 MB).
  char*  ws  = (char*)d_ws;
  size_t off = 0;
  auto carve = [&](size_t bytes) -> float* {
    float* p = (float*)(ws + off);
    off = (off + bytes + 255) & ~(size_t)255;
    return p;
  };
  float* cnt  = carve((size_t)N_NODES * 4);
  float* inv  = carve((size_t)N_NODES * 4);
  float* agg1 = carve((size_t)N_NODES * F_DIM * 4);
  float* h    = carve((size_t)N_NODES * F_DIM * 4);
  float* agg2 = carve((size_t)N_NODES * F_DIM * 4);

  float* out_ls = (float*)d_out;                         // [N, 64] log_softmax
  float* logits = (float*)d_out + (size_t)N_NODES * N_CLS;  // [N, 64] logits

  hipMemsetAsync(cnt,  0, (size_t)N_NODES * 4, stream);
  hipMemsetAsync(agg1, 0, (size_t)N_NODES * F_DIM * 4, stream);
  hipMemsetAsync(agg2, 0, (size_t)N_NODES * F_DIM * 4, stream);

  const int edgeBlocks = (N_EDGES + 7) / 8;   // 8 waves (edges) per 256-thr block
  const int mtiles     = N_NODES / 16;        // 3125 exactly
  const int gemmBlocks = (mtiles + 7) / 8;
  const int nodeBlocks = (N_NODES + 7) / 8;

  // Layer 1: aggregate, degrees, fused GEMM + bias + ReLU -> h
  scatter_accum_kernel<<<edgeBlocks, 256, 0, stream>>>(x, src, dst, agg1, cnt,
                                                       N_EDGES);
  inv_deg_kernel<<<(N_NODES + 255) / 256, 256, 0, stream>>>(cnt, inv, N_NODES);
  sage_gemm_kernel<8, true><<<gemmBlocks, 256, 0, stream>>>(
      agg1, inv, x, w1l, w1r, b1l, h, mtiles);

  // Layer 2: aggregate h (same degrees), fused GEMM + bias -> logits
  scatter_accum_kernel<<<edgeBlocks, 256, 0, stream>>>(h, src, dst, agg2,
                                                       nullptr, N_EDGES);
  sage_gemm_kernel<4, false><<<gemmBlocks, 256, 0, stream>>>(
      agg2, inv, h, w2l, w2r, b2l, logits, mtiles);

  // log_softmax into first half of d_out
  logsoftmax_kernel<<<nodeBlocks, 256, 0, stream>>>(logits, out_ls, N_NODES);
}
